// GroupWiseCorrelationCostVolume_30734785970549
// MI455X (gfx1250) — matrix operand order repaired and checked by reading, not compiled, not run on hardware
//
#include <hip/hip_runtime.h>
#include <hip/hip_bf16.h>

typedef __bf16 bf16_t;
typedef __attribute__((ext_vector_type(16))) __bf16 v16bf;
typedef __attribute__((ext_vector_type(8)))  __bf16 v8bf;
typedef __attribute__((ext_vector_type(8)))  float  v8f;

#define D_   48
#define H_   64
#define W_   128
#define C_   256
#define G_   32
#define CG_  8
#define HID_ 32
#define TAPS 27
#define NEG_SLOPE 0.2f

#define ACT_ELEMS ((size_t)D_*H_*W_*HID_)   // 12,582,912 bf16 elems = 25.2 MB
#define WR_ELEMS  (TAPS*2*32*16)            // 27,648 bf16 per layer (reordered)
#define ROW_POS   132                       // [-2..129]; halo zeros at idx 1 & 130

// ---------------------------------------------------------------------------
// CDNA5 async copy: 32 bytes/lane global -> LDS, tracked by ASYNCcnt.
// Per CDNA5 ISA the instruction IOFFSET is added to BOTH the LDS
// and global address, so the second 16B beat reuses the same base registers.
// ---------------------------------------------------------------------------
__device__ __forceinline__ void async_copy32(unsigned lds_addr, const void* gptr) {
    unsigned long long ga = (unsigned long long)(size_t)gptr;
    asm volatile("global_load_async_to_lds_b128 %0, %1, off"
                 :: "v"(lds_addr), "v"(ga) : "memory");
    asm volatile("global_load_async_to_lds_b128 %0, %1, off offset:16"
                 :: "v"(lds_addr), "v"(ga) : "memory");
}
__device__ __forceinline__ void wait_async0() {
    asm volatile("s_wait_asynccnt 0x0" ::: "memory");
}

// ---------------------------------------------------------------------------
// Reorder conv weights [O=32][I=32][3][3][3] f32 -> per-lane WMMA A-fragment
// layout, bf16:  wr[layer][tap][half][lane][slot]
//   M (out row)  = (lane&15) + 16*half
//   K (in chan)  : contiguous layout: lanes 0-15 hold K=0..15;
//                  lanes 16-31 hold K=16..31
// ---------------------------------------------------------------------------
__global__ void reorder_weights_k(const float* __restrict__ w0,
                                  const float* __restrict__ w1,
                                  const float* __restrict__ w2,
                                  const float* __restrict__ w3,
                                  bf16_t* __restrict__ wr) {
    int t = blockIdx.x * blockDim.x + threadIdx.x;
    if (t >= 4 * WR_ELEMS) return;
    int layer = t / WR_ELEMS;
    int r     = t - layer * WR_ELEMS;        // ((tap*2+half)*32+lane)*16+slot
    int slot  = r & 15;
    int lane  = (r >> 4) & 31;
    int half  = (r >> 9) & 1;
    int tap   = r >> 10;
    const float* src = (layer == 0) ? w0 : (layer == 1) ? w1 : (layer == 2) ? w2 : w3;
    int outc = (lane & 15) + 16 * half;
    int hi   = lane >> 4;
    int grp8 = (hi << 1) | (slot >> 3);
    int k    = (grp8 << 3) | (slot & 7);
    wr[t] = (bf16_t)src[(outc * HID_ + k) * TAPS + tap];
}

__global__ void cast_wf_k(const float* __restrict__ wf, bf16_t* __restrict__ out) {
    int t = blockIdx.x * blockDim.x + threadIdx.x;
    if (t < HID_ * TAPS) out[t] = (bf16_t)wf[t];
}

// ---------------------------------------------------------------------------
// Group-wise correlation volume -> channels-last bf16 [D][H][W][G]
// One block per (d,h), one thread per w. Inputs are L2-resident (16 MB).
// ---------------------------------------------------------------------------
__global__ __launch_bounds__(W_)
void gwc_volume_k(const float* __restrict__ L, const float* __restrict__ R,
                  bf16_t* __restrict__ vol) {
    int d = blockIdx.x >> 6;         // 0..47
    int h = blockIdx.x & (H_ - 1);   // 0..63
    int w = threadIdx.x;             // 0..127
    bf16_t* o = vol + (((size_t)d * H_ + h) * W_ + w) * G_;
    if (w < d) {
        #pragma unroll
        for (int g = 0; g < G_; ++g) o[g] = (bf16_t)0.f;
        return;
    }
    const float* lp = L + (size_t)h * W_ + w;
    const float* rp = R + (size_t)h * W_ + (w - d);
    for (int g = 0; g < G_; ++g) {
        float s = 0.f;
        #pragma unroll
        for (int c = 0; c < CG_; ++c) {
            size_t off = (size_t)(g * CG_ + c) * (H_ * W_);
            s += lp[off] * rp[off];
        }
        o[g] = (bf16_t)(s * (1.f / CG_));
    }
}

// ---------------------------------------------------------------------------
// 3x3x3 conv3d, 32->32 ch, SAME padding, leaky-ReLU. Implicit GEMM:
// block = 8 waves = one (d,h) row; each wave owns a 16(w) x 32(outC) tile.
// Per (kd,kh): async-stage the input row (+/-1 halo pre-zeroed) into LDS,
// then 3 kw shifts read B via ds_load and issue 2x v_wmma_f32_16x16x32_bf16.
// All padding skips are block-uniform => EXEC all-ones at every WMMA/barrier.
// ---------------------------------------------------------------------------
__global__ __launch_bounds__(256)
void conv3d_wmma_k(const bf16_t* __restrict__ act, const bf16_t* __restrict__ wr,
                   const float* __restrict__ bias, bf16_t* __restrict__ out) {
    __shared__ bf16_t sw[WR_ELEMS];        // 55,296 B weight fragments
    __shared__ bf16_t srow[ROW_POS * HID_]; // 8,448 B staged row
    {
        const unsigned int* s = (const unsigned int*)wr;
        unsigned int* dsh = (unsigned int*)sw;
        for (int i = threadIdx.x; i < WR_ELEMS / 2; i += 256) dsh[i] = s[i];
        // zero the halo entries once: logical w=-1 (idx 1), w=128 (idx 130)
        if (threadIdx.x < 16) {
            uint2 z; z.x = 0u; z.y = 0u;
            uint2* zp = (uint2*)srow;      // 8B units; 64B per position
            int t = threadIdx.x;
            zp[(t < 8) ? (8 + t) : (1040 + (t - 8))] = z;
        }
    }
    __syncthreads();

    int lane = threadIdx.x & 31;
    int wave = threadIdx.x >> 5;
    int h = blockIdx.x & (H_ - 1);
    int d = blockIdx.x >> 6;
    int n  = lane & 15;                 // output position within tile (N)
    int hi = lane >> 4;                 // channel half for B/K striping
    int w  = wave * 16 + n;             // this lane's output w (0..127)

    // staging slice for this thread: 256 threads x 32B cover 128 pos x 64B
    int pos   = threadIdx.x >> 1;       // 0..127
    int halfq = threadIdx.x & 1;        // which 32B half of the 64B position
    unsigned lrow = (unsigned)(size_t)(srow + (pos + 2) * HID_) + halfq * 32u;

    v8f c0 = {}; v8f c1 = {};
    for (int kd = 0; kd < 3; ++kd) {
        int dd = d + kd - 1;
        if (dd < 0 || dd >= D_) continue;           // block-uniform
        for (int kh = 0; kh < 3; ++kh) {
            int hh = h + kh - 1;
            if (hh < 0 || hh >= H_) continue;       // block-uniform
            __syncthreads();   // all waves done reading previous staged row
            const bf16_t* g = act + (((size_t)dd * H_ + hh) * W_ + pos) * HID_
                              + halfq * 16;
            async_copy32(lrow, g);
            wait_async0();
            __syncthreads();   // staged row visible to all waves
            #pragma unroll
            for (int kw = 0; kw < 3; ++kw) {
                int tap = (kd * 3 + kh) * 3 + kw;
                // logical position w+kw-1 lives at physical index w+kw+1
                v16bf b  = *(const v16bf*)(srow + (w + kw + 1) * HID_ + hi * 16);
                v16bf a0 = *(const v16bf*)(sw + ((tap * 2 + 0) * 32 + lane) * 16);
                v16bf a1 = *(const v16bf*)(sw + ((tap * 2 + 1) * 32 + lane) * 16);
                c0 = __builtin_amdgcn_wmma_f32_16x16x32_bf16(false, a0, false, b,
                                                             (short)0, c0, false, false);
                c1 = __builtin_amdgcn_wmma_f32_16x16x32_bf16(false, a1, false, b,
                                                             (short)0, c1, false, false);
            }
        }
    }

    // D layout: lane L holds N=L&15; VGPR r holds M = r + 8*(L>>4)
    bf16_t* op = out + (((size_t)d * H_ + h) * W_ + wave * 16 + n) * HID_;
    v8bf o0, o1;
    #pragma unroll
    for (int r = 0; r < 8; ++r) {
        float f0 = c0[r] + bias[r + 8 * hi];
        float f1 = c1[r] + bias[16 + r + 8 * hi];
        f0 = f0 > 0.f ? f0 : NEG_SLOPE * f0;
        f1 = f1 > 0.f ? f1 : NEG_SLOPE * f1;
        o0[r] = (bf16_t)f0;
        o1[r] = (bf16_t)f1;
    }
    *(v8bf*)(op + 8 * hi)      = o0;   // channels [8*hi .. 8*hi+7]
    *(v8bf*)(op + 16 + 8 * hi) = o1;   // channels [16+8*hi ..]
}

// ---------------------------------------------------------------------------
// Final 3x3x3 conv, 32 -> 1 channel, f32 output. Tiny (0.7 GFLOP): plain VALU.
// ---------------------------------------------------------------------------
__global__ __launch_bounds__(256)
void conv3d_final_k(const bf16_t* __restrict__ act, const bf16_t* __restrict__ wf,
                    const float* __restrict__ bfin, float* __restrict__ out) {
    __shared__ bf16_t swf[HID_ * TAPS];
    for (int i = threadIdx.x; i < HID_ * TAPS; i += 256) swf[i] = wf[i];
    __syncthreads();
    int v = blockIdx.x * 256 + threadIdx.x;   // 0..393215
    int w = v & (W_ - 1);
    int h = (v >> 7) & (H_ - 1);
    int d = v >> 13;
    float s = bfin[0];
    for (int kd = 0; kd < 3; ++kd) {
        int dd = d + kd - 1; if (dd < 0 || dd >= D_) continue;
        for (int kh = 0; kh < 3; ++kh) {
            int hh = h + kh - 1; if (hh < 0 || hh >= H_) continue;
            for (int kw = 0; kw < 3; ++kw) {
                int ww = w + kw - 1; if (ww < 0 || ww >= W_) continue;
                int tap = (kd * 3 + kh) * 3 + kw;
                const bf16_t* p = act + (((size_t)dd * H_ + hh) * W_ + ww) * HID_;
                #pragma unroll
                for (int c = 0; c < HID_; ++c)
                    s += (float)p[c] * (float)swf[c * TAPS + tap];
            }
        }
    }
    out[v] = s;
}

extern "C" void kernel_launch(void* const* d_in, const int* in_sizes, int n_in,
                              void* d_out, int out_size, void* d_ws, size_t ws_size,
                              hipStream_t stream) {
    const float* left  = (const float*)d_in[0];
    const float* right = (const float*)d_in[1];
    const float* w0 = (const float*)d_in[2];  const float* b0 = (const float*)d_in[3];
    const float* w1 = (const float*)d_in[4];  const float* b1 = (const float*)d_in[5];
    const float* w2 = (const float*)d_in[6];  const float* b2 = (const float*)d_in[7];
    const float* w3 = (const float*)d_in[8];  const float* b3 = (const float*)d_in[9];
    const float* wf = (const float*)d_in[10]; const float* bf = (const float*)d_in[11];
    (void)in_sizes; (void)n_in; (void)out_size; (void)ws_size;

    char* ws = (char*)d_ws;
    bf16_t* actA = (bf16_t*)ws;                                   // 25.2 MB
    bf16_t* actB = (bf16_t*)(ws + ACT_ELEMS * 2);                 // 25.2 MB
    bf16_t* wr   = (bf16_t*)(ws + ACT_ELEMS * 4);                 // 4 x 55 KB
    bf16_t* wfb  = (bf16_t*)(ws + ACT_ELEMS * 4 + (size_t)4 * WR_ELEMS * 2);

    hipLaunchKernelGGL(reorder_weights_k, dim3((4 * WR_ELEMS + 255) / 256), dim3(256),
                       0, stream, w0, w1, w2, w3, wr);
    hipLaunchKernelGGL(cast_wf_k, dim3(4), dim3(256), 0, stream, wf, wfb);
    hipLaunchKernelGGL(gwc_volume_k, dim3(D_ * H_), dim3(W_), 0, stream,
                       left, right, actA);

    hipLaunchKernelGGL(conv3d_wmma_k, dim3(D_ * H_), dim3(256), 0, stream,
                       actA, wr + (size_t)0 * WR_ELEMS, b0, actB);
    hipLaunchKernelGGL(conv3d_wmma_k, dim3(D_ * H_), dim3(256), 0, stream,
                       actB, wr + (size_t)1 * WR_ELEMS, b1, actA);
    hipLaunchKernelGGL(conv3d_wmma_k, dim3(D_ * H_), dim3(256), 0, stream,
                       actA, wr + (size_t)2 * WR_ELEMS, b2, actB);
    hipLaunchKernelGGL(conv3d_wmma_k, dim3(D_ * H_), dim3(256), 0, stream,
                       actB, wr + (size_t)3 * WR_ELEMS, b3, actA);

    hipLaunchKernelGGL(conv3d_final_k, dim3((D_ * H_ * W_) / 256), dim3(256),
                       0, stream, actA, wfb, bf, (float*)d_out);
}
